// LSTMGenerator_Diffusion_61787399520789
// MI455X (gfx1250) — matrix operand order repaired and checked by reading, not compiled, not run on hardware
//
#include <hip/hip_runtime.h>
#include <hip/hip_bf16.h>
#include <math.h>
#include <stdint.h>

typedef __bf16 bf16_t;
typedef __attribute__((ext_vector_type(16))) __bf16 v16bf;
typedef __attribute__((ext_vector_type(8)))  float  v8f;
typedef int v4i_t __attribute__((vector_size(16)));   // pointee type of async-LDS builtin

#define AS1 __attribute__((address_space(1)))
#define AS3 __attribute__((address_space(3)))

#define BATCH 256
#define HID   1024
#define GDIM  4096
#define XDIM  768          // (D-1)*L
#define KC0   1792         // 768 + 1024
#define KC1   2048         // 1024 + 1024
#define STEPS 100
#define OUTSL (BATCH*XDIM) // 196608 floats per diffusion slice

#if __has_builtin(__builtin_amdgcn_global_load_async_to_lds_b128)
#define HAVE_ASYNC_LDS 1
#else
#define HAVE_ASYNC_LDS 0
#endif

__device__ __forceinline__ void wait_async0() {
#if __has_builtin(__builtin_amdgcn_s_wait_asynccnt)
  __builtin_amdgcn_s_wait_asynccnt(0);
#else
  asm volatile("s_wait_asynccnt 0x0" ::: "memory");
#endif
}

// ---------------------------------------------------------------------------
// Generic bf16 WMMA GEMM:  C[M,N] = X[M,K] @ W[N,K]^T + bias[N]
// Block tile 64x64, 128 threads (4 waves), wave tile 32x32 (2x2 WMMA tiles).
// K staged 64 wide through double-buffered LDS with async copies
// (GLOBAL_LOAD_ASYNC_TO_LDS_B128, ASYNCcnt) overlapping WMMA compute; the
// final K chunk is peeled so the steady-state loop is branch-free.
// M,N,K multiples of 64.
// ---------------------------------------------------------------------------
#define TM 64
#define TN 64
#define TK 64
#define LDP 72   // LDS row stride in bf16 elems (144B, 16B-aligned)

union Frag { v16bf v; float4 f4[2]; };

__global__ __launch_bounds__(128)
void gemm_bias_bf16(const bf16_t* __restrict__ X, const bf16_t* __restrict__ W,
                    const float* __restrict__ bias, float* __restrict__ C,
                    int M, int N, int K)
{
  __shared__ bf16_t lA[2][TM * LDP];
  __shared__ bf16_t lB[2][TN * LDP];
  const int tid  = threadIdx.x;
  const int lane = tid & 31;
  const int wid  = tid >> 5;      // 0..3
  const int wm   = wid >> 1;      // wave row (32 rows each)
  const int wn   = wid & 1;       // wave col (32 cols each)
  const int bm   = blockIdx.y * TM;
  const int bn   = blockIdx.x * TN;
  const int half = lane >> 4;
  const int l16  = lane & 15;

  // per-thread staging coordinates: 4 chunks of 8 bf16 for A and for B
  int rr[4], cc[4];
  #pragma unroll
  for (int i = 0; i < 4; ++i) { int ch = tid + i*128; rr[i] = ch >> 3; cc[i] = (ch & 7) << 3; }

  const v8f zero = {0.f,0.f,0.f,0.f,0.f,0.f,0.f,0.f};
  v8f acc[2][2];
  acc[0][0]=zero; acc[0][1]=zero; acc[1][0]=zero; acc[1][1]=zero;

  // fragment compute on LDS buffer `bufi` (2 kc sub-steps of K=32)
  auto compute = [&](int bufi) {
    #pragma unroll
    for (int kc = 0; kc < 2; ++kc) {
      Frag a[2], b[2];
      // A fragment: lane l holds row m=l16; elems 0..7 = K half*8+0..7,
      // elems 8..15 = K 16+half*8+0..7 (CDNA5 16-bit A layout)
      #pragma unroll
      for (int mt = 0; mt < 2; ++mt) {
        int r  = wm*32 + mt*16 + l16;
        int cb = kc*32 + half*8;
        a[mt].f4[0] = *(const float4*)&lA[bufi][r*LDP + cb];
        a[mt].f4[1] = *(const float4*)&lA[bufi][r*LDP + cb + 16];
      }
      // B fragment: lane l holds col n=l16; elems 0..7 = K half*8+0..7, elems 8..15 = K 16+half*8+0..7
      #pragma unroll
      for (int nt = 0; nt < 2; ++nt) {
        int r  = wn*32 + nt*16 + l16;
        int cb = kc*32 + half*8;
        b[nt].f4[0] = *(const float4*)&lB[bufi][r*LDP + cb];
        b[nt].f4[1] = *(const float4*)&lB[bufi][r*LDP + cb + 16];
      }
      #pragma unroll
      for (int mt = 0; mt < 2; ++mt)
        #pragma unroll
        for (int nt = 0; nt < 2; ++nt)
          acc[mt][nt] = __builtin_amdgcn_wmma_f32_16x16x32_bf16(
              false, a[mt].v, false, b[nt].v, (short)0, acc[mt][nt], false, false);
    }
  };

#if HAVE_ASYNC_LDS
  #define STAGE(bufi, kbase)                                                        \
    _Pragma("unroll")                                                               \
    for (int i = 0; i < 4; ++i) {                                                   \
      __builtin_amdgcn_global_load_async_to_lds_b128(                               \
          (AS1 v4i_t*)(uintptr_t)&X[(size_t)(bm + rr[i])*K + (kbase) + cc[i]],      \
          (AS3 v4i_t*)&lA[bufi][rr[i]*LDP + cc[i]], 0, 0);                          \
      __builtin_amdgcn_global_load_async_to_lds_b128(                               \
          (AS1 v4i_t*)(uintptr_t)&W[(size_t)(bn + rr[i])*K + (kbase) + cc[i]],      \
          (AS3 v4i_t*)&lB[bufi][rr[i]*LDP + cc[i]], 0, 0);                          \
    }

  STAGE(0, 0)
  int buf = 0;
  int kb = 0;
  for (; kb + TK < K; kb += TK, buf ^= 1) {   // steady state: branch-free body
    wait_async0();        // copies into `buf` complete (this wave)
    __syncthreads();      // all waves: copies visible, prev compute done
    STAGE(buf ^ 1, kb + TK)                   // overlap next copies with compute
    compute(buf);
  }
  wait_async0();                               // peeled final chunk
  __syncthreads();
  compute(buf);
#else
  float4 ra[4], rb[4];
  #define LOADREGS(kbase)                                                          \
    _Pragma("unroll")                                                              \
    for (int i = 0; i < 4; ++i) {                                                  \
      ra[i] = *(const float4*)&X[(size_t)(bm + rr[i])*K + (kbase) + cc[i]];        \
      rb[i] = *(const float4*)&W[(size_t)(bn + rr[i])*K + (kbase) + cc[i]];        \
    }
  #define STOREREGS(bufi)                                                          \
    _Pragma("unroll")                                                              \
    for (int i = 0; i < 4; ++i) {                                                  \
      *(float4*)&lA[bufi][rr[i]*LDP + cc[i]] = ra[i];                              \
      *(float4*)&lB[bufi][rr[i]*LDP + cc[i]] = rb[i];                              \
    }
  LOADREGS(0)
  int buf = 0;
  int kb = 0;
  for (; kb + TK < K; kb += TK, buf ^= 1) {
    STOREREGS(buf)
    __syncthreads();
    LOADREGS(kb + TK)                          // overlap with compute
    compute(buf);
  }
  STOREREGS(buf)
  __syncthreads();
  compute(buf);
#endif

  // epilogue: C/D layout -> lane l: col n=l16, rows half*8 + r
  #pragma unroll
  for (int mt = 0; mt < 2; ++mt)
    #pragma unroll
    for (int nt = 0; nt < 2; ++nt) {
      int n = bn + wn*32 + nt*16 + l16;
      float bv = bias[n];
      int mbase = bm + wm*32 + mt*16 + half*8;
      #pragma unroll
      for (int r = 0; r < 8; ++r)
        C[(size_t)(mbase + r)*N + n] = acc[mt][nt][r] + bv;
    }
}

// ---------------------------------------------------------------------------
// Elementwise / setup kernels
// ---------------------------------------------------------------------------
__device__ __forceinline__ float sigf(float x) { return 1.f / (1.f + expf(-x)); }

__global__ void concat_w_bf16(const float* __restrict__ WA, int KA,
                              const float* __restrict__ WB, int KB,
                              bf16_t* __restrict__ dst, int total)
{
  int idx = blockIdx.x * 256 + threadIdx.x;
  if (idx >= total) return;
  int Kt = KA + KB;
  int n = idx / Kt, k = idx - n * Kt;
  float v = (k < KA) ? WA[(size_t)n * KA + k] : WB[(size_t)n * KB + (k - KA)];
  dst[idx] = (bf16_t)v;
}

__global__ void f32_to_bf16_k(const float* __restrict__ s, bf16_t* __restrict__ d, int n)
{
  int i = blockIdx.x * 256 + threadIdx.x;
  if (i < n) d[i] = (bf16_t)s[i];
}

__global__ void vec_add_k(const float* a, const float* b, float* d, int n)
{
  int i = blockIdx.x * 256 + threadIdx.x;
  if (i < n) d[i] = a[i] + b[i];
}

__global__ void fill_zero_k(float* p, int n)
{
  int i = blockIdx.x * 256 + threadIdx.x;
  if (i < n) p[i] = 0.f;
}

// h1 = relu(n0 @ A1^T + b1) ; n0 (256,4), A1 (1024,4) -> (256,1024)
__global__ void resnet_l1_k(const float* __restrict__ n0, const float* __restrict__ A1,
                            const float* __restrict__ b1,
                            float* __restrict__ h1f, bf16_t* __restrict__ h1b)
{
  int idx = blockIdx.x * 256 + threadIdx.x;   // 256*1024
  int m = idx >> 10, n = idx & 1023;
  float s = b1[n];
  #pragma unroll
  for (int k = 0; k < 4; ++k) s += n0[m*4+k] * A1[n*4+k];
  s = fmaxf(s, 0.f);
  h1f[idx] = s;
  h1b[idx] = (bf16_t)s;
}

// h2 = relu(pre) + h1 -> bf16
__global__ void relu_add_k(const float* __restrict__ pre, const float* __restrict__ h1f,
                           bf16_t* __restrict__ h2b)
{
  int idx = blockIdx.x * 256 + threadIdx.x;   // 256*1024
  h2b[idx] = (bf16_t)(fmaxf(pre[idx], 0.f) + h1f[idx]);
}

// tanh of (256,2048) pre, split cols [0:1024) -> dst0 (bf16, strided),
// [1024:2048) -> dst1 (bf16, strided)
__global__ void tanh_split_bf16_k(const float* __restrict__ pre,
                                  bf16_t* __restrict__ dst0, int ld0,
                                  bf16_t* __restrict__ dst1, int ld1)
{
  int idx = blockIdx.x * 256 + threadIdx.x;   // 256*2048
  int m = idx >> 11, n = idx & 2047;
  float t = tanhf(pre[idx]);
  if (n < 1024) dst0[(size_t)m * ld0 + n] = (bf16_t)t;
  else          dst1[(size_t)m * ld1 + (n - 1024)] = (bf16_t)t;
}

// tanh of (256,2048) pre, split into two contiguous f32 (256,1024) buffers
__global__ void tanh_split_f32_k(const float* __restrict__ pre,
                                 float* __restrict__ c0, float* __restrict__ c1)
{
  int idx = blockIdx.x * 256 + threadIdx.x;   // 256*2048
  int m = idx >> 11, n = idx & 2047;
  float t = tanhf(pre[idx]);
  if (n < 1024) c0[m*1024 + n] = t;
  else          c1[m*1024 + (n - 1024)] = t;
}

// LSTM cell elementwise: gates (256,4096) f32, c (256,1024) f32 in/out,
// h written bf16 to two strided destinations; optional tanh(h) bf16 out.
__global__ void lstm_cell_k(const float* __restrict__ gates, float* __restrict__ c,
                            bf16_t* __restrict__ hA, int ldA,
                            bf16_t* __restrict__ hB, int ldB,
                            bf16_t* __restrict__ tanh_dst)
{
  int idx = blockIdx.x * 256 + threadIdx.x;   // 256*1024
  int m = idx >> 10, n = idx & 1023;
  const float* g = gates + (size_t)m * GDIM;
  float gi = g[n], gf = g[n + 1024], gg = g[n + 2048], go = g[n + 3072];
  float cn = sigf(gf) * c[idx] + sigf(gi) * tanhf(gg);
  c[idx] = cn;
  float h = sigf(go) * tanhf(cn);
  bf16_t hb = (bf16_t)h;
  hA[(size_t)m * ldA + n] = hb;
  hB[(size_t)m * ldB + n] = hb;
  if (tanh_dst) tanh_dst[idx] = (bf16_t)tanhf(h);
}

// Diffusion update: dec = (prev - (1-a)/sqrt(1-ba)*raw)/sqrt(a) [+ sqrt(be)*noise]
__global__ void diffusion_k(const float* __restrict__ raw, float* __restrict__ prev,
                            const float* __restrict__ alphas, const float* __restrict__ betas,
                            const float* __restrict__ baral, const float* __restrict__ sn,
                            int t, float* __restrict__ outslice, bf16_t* __restrict__ xcat0)
{
  int idx = blockIdx.x * 256 + threadIdx.x;   // 256*768
  int m = idx / XDIM, r = idx - m * XDIM;
  int id = STEPS - t;
  float a = alphas[id], ba = baral[id], be = betas[id];
  float d = (prev[idx] - (1.f - a) * rsqrtf(1.f - ba) * raw[idx]) * rsqrtf(a);
  if (t < STEPS - 1) d += sqrtf(be) * sn[t * BATCH + m];
  outslice[idx] = d;
  prev[idx] = d;
  xcat0[(size_t)m * KC0 + r] = (bf16_t)d;
}

// prev0 = z[:,:,:3]; also seed xcat0 and the last output slice
__global__ void init_prev_k(const float* __restrict__ z, float* __restrict__ prev,
                            bf16_t* __restrict__ xcat0, float* __restrict__ outlast)
{
  int idx = blockIdx.x * 256 + threadIdx.x;   // 256*768
  int b = idx / XDIM, r = idx - b * XDIM;
  int l = r / 3, d = r - l * 3;
  float v = z[(size_t)b * 1024 + l * 4 + d];  // z is (256,256,4)
  prev[idx] = v;
  xcat0[(size_t)b * KC0 + r] = (bf16_t)v;
  outlast[idx] = v;
}

// ---------------------------------------------------------------------------
extern "C" void kernel_launch(void* const* d_in, const int* in_sizes, int n_in,
                              void* d_out, int out_size, void* d_ws, size_t ws_size,
                              hipStream_t stream)
{
  (void)in_sizes; (void)n_in; (void)out_size; (void)ws_size;
  const float* z     = (const float*)d_in[4];
  const float* n0    = (const float*)d_in[5];
  const float* sn    = (const float*)d_in[6];
  const float* alphas= (const float*)d_in[7];
  const float* betas = (const float*)d_in[8];
  const float* baral = (const float*)d_in[9];
  const float* A1a = (const float*)d_in[10]; const float* b1a = (const float*)d_in[11];
  const float* A2a = (const float*)d_in[12]; const float* b2a = (const float*)d_in[13];
  const float* A3a = (const float*)d_in[14]; const float* b3a = (const float*)d_in[15];
  const float* A1b = (const float*)d_in[16]; const float* b1b = (const float*)d_in[17];
  const float* A2b = (const float*)d_in[18]; const float* b2b = (const float*)d_in[19];
  const float* A3b = (const float*)d_in[20]; const float* b3b = (const float*)d_in[21];
  const float* Wih0= (const float*)d_in[22]; const float* Whh0= (const float*)d_in[23];
  const float* bih0= (const float*)d_in[24]; const float* bhh0= (const float*)d_in[25];
  const float* Wih1= (const float*)d_in[26]; const float* Whh1= (const float*)d_in[27];
  const float* bih1= (const float*)d_in[28]; const float* bhh1= (const float*)d_in[29];
  const float* Wlin= (const float*)d_in[30];
  float* out = (float*)d_out;

  // workspace carve-up (256B aligned)
  char* wsp = (char*)d_ws;
  auto alloc = [&](size_t bytes) {
    void* p = (void*)wsp;
    wsp += (bytes + 255) & ~(size_t)255;
    return p;
  };
  bf16_t* wcat0 = (bf16_t*)alloc((size_t)GDIM * KC0 * 2);   // [Wih0|Whh0] bf16
  bf16_t* wcat1 = (bf16_t*)alloc((size_t)GDIM * KC1 * 2);   // [Wih1|Whh1] bf16
  bf16_t* wlinb = (bf16_t*)alloc((size_t)XDIM * HID * 2);
  bf16_t* A2ab  = (bf16_t*)alloc((size_t)HID * HID * 2);
  bf16_t* A3ab  = (bf16_t*)alloc((size_t)2 * HID * HID * 2);
  bf16_t* A2bb  = (bf16_t*)alloc((size_t)HID * HID * 2);
  bf16_t* A3bb  = (bf16_t*)alloc((size_t)2 * HID * HID * 2);
  float*  bsum0 = (float*)alloc(GDIM * 4);
  float*  bsum1 = (float*)alloc(GDIM * 4);
  float*  zbias = (float*)alloc(GDIM * 4);
  float*  gates = (float*)alloc((size_t)BATCH * GDIM * 4);  // also resnet pre-act scratch
  bf16_t* xcat0 = (bf16_t*)alloc((size_t)BATCH * KC0 * 2);  // [x | h0]
  bf16_t* xcat1 = (bf16_t*)alloc((size_t)BATCH * KC1 * 2);  // [h0n | h1]
  float*  c0    = (float*)alloc((size_t)BATCH * HID * 4);
  float*  c1    = (float*)alloc((size_t)BATCH * HID * 4);
  float*  h1f   = (float*)alloc((size_t)BATCH * HID * 4);
  bf16_t* h1bb  = (bf16_t*)alloc((size_t)BATCH * HID * 2);
  bf16_t* h2bb  = (bf16_t*)alloc((size_t)BATCH * HID * 2);
  bf16_t* tanhb = (bf16_t*)alloc((size_t)BATCH * HID * 2);
  float*  decraw= (float*)alloc((size_t)BATCH * XDIM * 4);
  float*  prev  = (float*)alloc((size_t)BATCH * XDIM * 4);

  const int T = 256;
  auto blks = [](size_t n) { return (unsigned)((n + 255) / 256); };

  // ---- weight prep (captured into the graph; trivial vs GEMM work) ----
  concat_w_bf16<<<blks((size_t)GDIM*KC0), T, 0, stream>>>(Wih0, XDIM, Whh0, HID, wcat0, GDIM*KC0);
  concat_w_bf16<<<blks((size_t)GDIM*KC1), T, 0, stream>>>(Wih1, HID,  Whh1, HID, wcat1, GDIM*KC1);
  f32_to_bf16_k<<<blks((size_t)XDIM*HID), T, 0, stream>>>(Wlin, wlinb, XDIM*HID);
  f32_to_bf16_k<<<blks((size_t)HID*HID),  T, 0, stream>>>(A2a, A2ab, HID*HID);
  f32_to_bf16_k<<<blks((size_t)2*HID*HID),T, 0, stream>>>(A3a, A3ab, 2*HID*HID);
  f32_to_bf16_k<<<blks((size_t)HID*HID),  T, 0, stream>>>(A2b, A2bb, HID*HID);
  f32_to_bf16_k<<<blks((size_t)2*HID*HID),T, 0, stream>>>(A3b, A3bb, 2*HID*HID);
  vec_add_k<<<blks(GDIM), T, 0, stream>>>(bih0, bhh0, bsum0, GDIM);
  vec_add_k<<<blks(GDIM), T, 0, stream>>>(bih1, bhh1, bsum1, GDIM);
  fill_zero_k<<<blks(GDIM), T, 0, stream>>>(zbias, GDIM);

  // ---- ResNet A -> initial h states (into xcat0 h-part / xcat1 h1-part) ----
  resnet_l1_k<<<blks((size_t)BATCH*HID), T, 0, stream>>>(n0, A1a, b1a, h1f, h1bb);
  gemm_bias_bf16<<<dim3(HID/TN, BATCH/TM), 128, 0, stream>>>(h1bb, A2ab, b2a, gates, BATCH, HID, HID);
  relu_add_k<<<blks((size_t)BATCH*HID), T, 0, stream>>>(gates, h1f, h2bb);
  gemm_bias_bf16<<<dim3(2*HID/TN, BATCH/TM), 128, 0, stream>>>(h2bb, A3ab, b3a, gates, BATCH, 2*HID, HID);
  tanh_split_bf16_k<<<blks((size_t)BATCH*2*HID), T, 0, stream>>>(gates, xcat0 + XDIM, KC0, xcat1 + HID, KC1);

  // ---- ResNet B -> initial c states ----
  resnet_l1_k<<<blks((size_t)BATCH*HID), T, 0, stream>>>(n0, A1b, b1b, h1f, h1bb);
  gemm_bias_bf16<<<dim3(HID/TN, BATCH/TM), 128, 0, stream>>>(h1bb, A2bb, b2b, gates, BATCH, HID, HID);
  relu_add_k<<<blks((size_t)BATCH*HID), T, 0, stream>>>(gates, h1f, h2bb);
  gemm_bias_bf16<<<dim3(2*HID/TN, BATCH/TM), 128, 0, stream>>>(h2bb, A3bb, b3b, gates, BATCH, 2*HID, HID);
  tanh_split_f32_k<<<blks((size_t)BATCH*2*HID), T, 0, stream>>>(gates, c0, c1);

  // ---- prev0 = z[:,:,:3]; outputs[::-1] puts it at slice 99 ----
  init_prev_k<<<blks((size_t)OUTSL), T, 0, stream>>>(z, prev, xcat0, out + (size_t)(STEPS-1) * OUTSL);

  // ---- 99 sequential diffusion steps ----
  for (int t = 1; t < STEPS; ++t) {
    // layer 0: gates = [x|h0] @ [Wih0|Whh0]^T + (bih0+bhh0)
    gemm_bias_bf16<<<dim3(GDIM/TN, BATCH/TM), 128, 0, stream>>>(xcat0, wcat0, bsum0, gates, BATCH, GDIM, KC0);
    // h0n -> xcat1[:, :H] (layer-1 input) and xcat0[:, X:] (next-step state)
    lstm_cell_k<<<blks((size_t)BATCH*HID), T, 0, stream>>>(gates, c0, xcat1, KC1, xcat0 + XDIM, KC0, (bf16_t*)nullptr);
    // layer 1: gates = [h0n|h1] @ [Wih1|Whh1]^T + (bih1+bhh1)
    gemm_bias_bf16<<<dim3(GDIM/TN, BATCH/TM), 128, 0, stream>>>(xcat1, wcat1, bsum1, gates, BATCH, GDIM, KC1);
    // h1n -> xcat1[:, H:] (next-step state); tanh(h1n) -> linear-head input
    lstm_cell_k<<<blks((size_t)BATCH*HID), T, 0, stream>>>(gates, c1, xcat1 + HID, KC1, xcat1 + HID, KC1, tanhb);
    // dec_raw = tanh(h1n) @ Wlin^T
    gemm_bias_bf16<<<dim3(XDIM/TN, BATCH/TM), 128, 0, stream>>>(tanhb, wlinb, zbias, decraw, BATCH, XDIM, HID);
    // diffusion update -> out slice (reversed), prev carry, next x input
    diffusion_k<<<blks((size_t)OUTSL), T, 0, stream>>>(decraw, prev, alphas, betas, baral, sn, t,
                                                       out + (size_t)(STEPS-1-t) * OUTSL, xcat0);
  }
}
